// TriangleMultiplication_15564961481078
// MI455X (gfx1250) — matrix operand order repaired and checked by reading, not compiled, not run on hardware
//
#include <hip/hip_runtime.h>
#include <hip/hip_bf16.h>
#include <math.h>

// ---------------------------------------------------------------------------
// Triangle multiplication (outgoing), B=1, N=512, D=128, fp32 in/out.
// bf16 WMMA (v_wmma_f32_16x16x32_bf16) with f32 accumulation; K2 uses
// double-buffered global_load_async_to_lds_b128 staging (ASYNCcnt).
// ---------------------------------------------------------------------------

#define DD   128
#define NSZ  512
#define NROW (512*512)      // 262144 flattened rows
#define EPSV 1e-5f
#define KSTEP 32
#define LSTR  40            // padded LDS row stride (bf16 elems) = 80B, 16B-aligned

typedef __bf16 bf16_t;
typedef __attribute__((ext_vector_type(16))) __bf16 v16bf;
typedef __attribute__((ext_vector_type(8)))  __bf16 v8bf;
typedef __attribute__((ext_vector_type(8)))  float  v8f;
typedef __attribute__((ext_vector_type(4)))  float  v4f;

__device__ __forceinline__ float sigmoidf_(float x) {
    // v_exp_f32 + v_rcp_f32 (avoid the IEEE divide expansion)
    return __builtin_amdgcn_rcpf(1.0f + __expf(-x));
}

__device__ __forceinline__ v8f wmma_bf16(v16bf a, v16bf b, v8f c) {
    return __builtin_amdgcn_wmma_f32_16x16x32_bf16(false, a, false, b, (short)0, c,
                                                   false, false);
}

__device__ __forceinline__ v8f zero8() {
    v8f c;
#pragma unroll
    for (int v = 0; v < 8; ++v) c[v] = 0.0f;
    return c;
}

// B fragment (K=32 x N=16) from a bf16 row-major [128][128] weight matrix.
__device__ __forceinline__ v16bf load_bfrag_w(const bf16_t* __restrict__ W,
                                              int nt, int ks, int lane) {
    int e = nt * 16 + (lane & 15);
    int k = ks * 32 + ((lane >> 4) << 3);
    const v8bf* p = (const v8bf*)(W + (size_t)e * DD + k);
    v8bf lo = p[0], hi = p[2];
    v16bf r;
#pragma unroll
    for (int i = 0; i < 8; ++i) { r[i] = lo[i]; r[i + 8] = hi[i]; }
    return r;
}

// ---------------------------------------------------------------------------
// K0: fp32 -> bf16 weight conversion (6 matrices: ga, la, gb, lb, g, o)
// ---------------------------------------------------------------------------
__global__ void k0_cvt_weights(const float* __restrict__ Wga, const float* __restrict__ Wla,
                               const float* __restrict__ Wgb, const float* __restrict__ Wlb,
                               const float* __restrict__ Wg,  const float* __restrict__ Wo,
                               bf16_t* __restrict__ Wb) {
    int idx = blockIdx.x * blockDim.x + threadIdx.x;
    if (idx >= 6 * DD * DD) return;
    int m = idx / (DD * DD), e = idx % (DD * DD);
    const float* src = (m == 0) ? Wga : (m == 1) ? Wla : (m == 2) ? Wgb
                     : (m == 3) ? Wlb : (m == 4) ? Wg : Wo;
    Wb[idx] = (bf16_t)src[e];
}

// ---------------------------------------------------------------------------
// K1: fused LayerNorm + 5 projections + gating. la/lb written channel-major
// ([d][row]) via LDS transpose; gsig kept [row][d].
// ---------------------------------------------------------------------------
__global__ __launch_bounds__(128) void k1_ln_proj(
    const float* __restrict__ pair,
    const float* __restrict__ ng, const float* __restrict__ nb,
    const bf16_t* __restrict__ Wb,
    const float* __restrict__ b_ga, const float* __restrict__ b_la,
    const float* __restrict__ b_gb, const float* __restrict__ b_lb,
    const float* __restrict__ b_g,
    bf16_t* __restrict__ la_t, bf16_t* __restrict__ lb_t,
    float* __restrict__ gsig, int tiles_per_wave)
{
    __shared__ __align__(16) bf16_t xpose[4][DD * 16];   // per-wave [ch][row] tile

    const int lane = threadIdx.x & 31;
    const int wave = threadIdx.x >> 5;
    const int h    = lane >> 4;
    const int ln16 = lane & 15;

    for (int t = 0; t < tiles_per_wave; ++t) {
        const int tile    = (blockIdx.x * 4 + wave) * tiles_per_wave + t;
        const int rowbase = tile * 16;
        const int row     = rowbase + ln16;

        // ---- load x: 64 fp32 per lane in the A-fragment d-pattern ----
        float xv[64];
        float s = 0.0f, ss = 0.0f;
#pragma unroll
        for (int g = 0; g < 8; ++g) {
            const int db = (g >> 1) * 32 + (g & 1) * 16 + h * 8;
            v4f a = *(const v4f*)(pair + (size_t)row * DD + db);
            v4f b = *(const v4f*)(pair + (size_t)row * DD + db + 4);
#pragma unroll
            for (int e = 0; e < 4; ++e) {
                xv[g * 8 + e]     = a[e];
                xv[g * 8 + 4 + e] = b[e];
                s += a[e] + b[e];
                ss += a[e] * a[e] + b[e] * b[e];
            }
        }
        s  += __shfl_xor(s, 16);
        ss += __shfl_xor(ss, 16);
        const float mu   = s * (1.0f / 128.0f);
        const float var  = ss * (1.0f / 128.0f) - mu * mu;
        const float rstd = rsqrtf(var + EPSV);

        // ---- normalize * gamma + beta, pack bf16 A fragments ----
        v16bf A[4];
#pragma unroll
        for (int g = 0; g < 8; ++g) {
            const int db = (g >> 1) * 32 + (g & 1) * 16 + h * 8;
            v4f g0 = *(const v4f*)(ng + db);
            v4f g1 = *(const v4f*)(ng + db + 4);
            v4f c0 = *(const v4f*)(nb + db);
            v4f c1 = *(const v4f*)(nb + db + 4);
#pragma unroll
            for (int e = 0; e < 4; ++e) {
                float n0 = (xv[g * 8 + e]     - mu) * rstd * g0[e] + c0[e];
                float n1 = (xv[g * 8 + 4 + e] - mu) * rstd * g1[e] + c1[e];
                A[g >> 1][(g & 1) * 8 + e]     = (bf16_t)n0;
                A[g >> 1][(g & 1) * 8 + 4 + e] = (bf16_t)n1;
            }
        }

        // ---- gated projections ----
#pragma unroll 1
        for (int mp = 0; mp < 2; ++mp) {
            const bf16_t* Wgt = Wb + (size_t)(2 * mp) * DD * DD;
            const bf16_t* Wln = Wb + (size_t)(2 * mp + 1) * DD * DD;
            const float* bgv = mp ? b_gb : b_ga;
            const float* blv = mp ? b_lb : b_la;
            bf16_t* dst = mp ? lb_t : la_t;

#pragma unroll
            for (int nt = 0; nt < 8; ++nt) {
                v8f c1 = zero8(), c2 = zero8();
#pragma unroll
                for (int ks = 0; ks < 4; ++ks) {
                    c1 = wmma_bf16(A[ks], load_bfrag_w(Wgt, nt, ks, lane), c1);
                    c2 = wmma_bf16(A[ks], load_bfrag_w(Wln, nt, ks, lane), c2);
                }
                const float bb1 = bgv[nt * 16 + ln16];
                const float bb2 = blv[nt * 16 + ln16];
                v8bf pk;
#pragma unroll
                for (int v = 0; v < 8; ++v)
                    pk[v] = (bf16_t)(sigmoidf_(c1[v] + bb1) * (c2[v] + bb2));
                *(v8bf*)&xpose[wave][(nt * 16 + ln16) * 16 + h * 8] = pk;
            }
            __syncthreads();
#pragma unroll
            for (int it = 0; it < 4; ++it) {
                const int ch = it * 32 + lane;
                v8bf lo = *(const v8bf*)&xpose[wave][ch * 16];
                v8bf hi = *(const v8bf*)&xpose[wave][ch * 16 + 8];
                bf16_t* gp = dst + (size_t)ch * NROW + rowbase;
                *(v8bf*)gp       = lo;
                *(v8bf*)(gp + 8) = hi;
            }
            __syncthreads();
        }

        // ---- output gate projection ----
#pragma unroll
        for (int nt = 0; nt < 8; ++nt) {
            v8f c = zero8();
#pragma unroll
            for (int ks = 0; ks < 4; ++ks)
                c = wmma_bf16(A[ks], load_bfrag_w(Wb + (size_t)4 * DD * DD, nt, ks, lane), c);
            const float bb = b_g[nt * 16 + ln16];
#pragma unroll
            for (int v = 0; v < 8; ++v)
                gsig[(size_t)(rowbase + v + 8 * h) * DD + nt * 16 + ln16] =
                    sigmoidf_(c[v] + bb);
        }
    }
}

// ---------------------------------------------------------------------------
// K2: triangle contraction, 128 channel GEMMs P_d = LA_d * LB_d^T.
// Block: one channel, 128x128 tile. A/B K-slabs (128x32 bf16 each) staged in
// LDS via double-buffered global_load_async_to_lds_b128; waves read WMMA
// fragments with ds_load_b128. ASYNCcnt-synchronized.
// ---------------------------------------------------------------------------
__global__ __launch_bounds__(256) void k2_triangle(
    const bf16_t* __restrict__ la_t, const bf16_t* __restrict__ lb_t,
    float* __restrict__ p_t)
{
    __shared__ __align__(16) bf16_t Abuf[2][128 * LSTR];
    __shared__ __align__(16) bf16_t Bbuf[2][128 * LSTR];

    const int tid  = threadIdx.x;
    const int lane = tid & 31;
    const int wave = tid >> 5;
    const int wi = wave & 3, wj = wave >> 2;     // 4x2 wave grid, 32x64 per wave
    const int d  = blockIdx.z;
    const int ib = blockIdx.x * 128;
    const int jb = blockIdx.y * 128;
    const int ln16 = lane & 15, h = lane >> 4;

    const bf16_t* Am = la_t + (size_t)d * NROW;
    const bf16_t* Bm = lb_t + (size_t)d * NROW;

    // 128 rows x 4 x 16B chunks per slab; 256 threads move 2 chunks per matrix
    const int r0 = tid >> 2, cc = tid & 3;
    const int r1 = r0 + 64;

    auto issue = [&](int buf, int kb) {
        unsigned da0 = (unsigned)(size_t)&Abuf[buf][r0 * LSTR + cc * 8];
        unsigned da1 = (unsigned)(size_t)&Abuf[buf][r1 * LSTR + cc * 8];
        unsigned db0 = (unsigned)(size_t)&Bbuf[buf][r0 * LSTR + cc * 8];
        unsigned db1 = (unsigned)(size_t)&Bbuf[buf][r1 * LSTR + cc * 8];
        unsigned long long ga0 = (unsigned long long)(size_t)(Am + (size_t)(ib + r0) * NSZ + kb + cc * 8);
        unsigned long long ga1 = (unsigned long long)(size_t)(Am + (size_t)(ib + r1) * NSZ + kb + cc * 8);
        unsigned long long gb0 = (unsigned long long)(size_t)(Bm + (size_t)(jb + r0) * NSZ + kb + cc * 8);
        unsigned long long gb1 = (unsigned long long)(size_t)(Bm + (size_t)(jb + r1) * NSZ + kb + cc * 8);
        asm volatile("global_load_async_to_lds_b128 %0, %1, off" :: "v"(da0), "v"(ga0) : "memory");
        asm volatile("global_load_async_to_lds_b128 %0, %1, off" :: "v"(da1), "v"(ga1) : "memory");
        asm volatile("global_load_async_to_lds_b128 %0, %1, off" :: "v"(db0), "v"(gb0) : "memory");
        asm volatile("global_load_async_to_lds_b128 %0, %1, off" :: "v"(db1), "v"(gb1) : "memory");
    };

    v8f C[2][4];
#pragma unroll
    for (int ti = 0; ti < 2; ++ti)
#pragma unroll
        for (int tj = 0; tj < 4; ++tj) C[ti][tj] = zero8();

    int buf = 0;
    issue(0, 0);

    for (int kb = 0; kb < NSZ; kb += KSTEP) {
        const bool more = (kb + KSTEP) < NSZ;
        if (more) issue(buf ^ 1, kb + KSTEP);           // prefetch next slab
        if (more) asm volatile("s_wait_asynccnt 0x4" ::: "memory");  // current slab done
        else      asm volatile("s_wait_asynccnt 0x0" ::: "memory");
        __syncthreads();

        v16bf Af[2], Bf[4];
#pragma unroll
        for (int ti = 0; ti < 2; ++ti) {
            const bf16_t* rb = &Abuf[buf][(wi * 32 + ti * 16 + ln16) * LSTR];
            v8bf lo = *(const v8bf*)(rb + h * 8);        // K = h*8 + 0..7
            v8bf hi = *(const v8bf*)(rb + 16 + h * 8);   // K = 16 + h*8 + 0..7
#pragma unroll
            for (int i = 0; i < 8; ++i) { Af[ti][i] = lo[i]; Af[ti][i + 8] = hi[i]; }
        }
#pragma unroll
        for (int tj = 0; tj < 4; ++tj) {
            const bf16_t* rb = &Bbuf[buf][(wj * 64 + tj * 16 + ln16) * LSTR];
            v8bf lo = *(const v8bf*)(rb + h * 8);        // K = h*8 + 0..7
            v8bf hi = *(const v8bf*)(rb + 16 + h * 8);   // K = 16 + h*8 + 0..7
#pragma unroll
            for (int i = 0; i < 8; ++i) { Bf[tj][i] = lo[i]; Bf[tj][i + 8] = hi[i]; }
        }
#pragma unroll
        for (int ti = 0; ti < 2; ++ti)
#pragma unroll
            for (int tj = 0; tj < 4; ++tj)
                C[ti][tj] = wmma_bf16(Af[ti], Bf[tj], C[ti][tj]);

        __syncthreads();    // slab fully consumed before it is overwritten
        buf ^= 1;
    }

    float* Pm = p_t + (size_t)d * NROW;
#pragma unroll
    for (int ti = 0; ti < 2; ++ti)
#pragma unroll
        for (int tj = 0; tj < 4; ++tj)
#pragma unroll
            for (int v = 0; v < 8; ++v)
                Pm[(size_t)(ib + wi * 32 + ti * 16 + v + 8 * h) * NSZ +
                   jb + wj * 64 + tj * 16 + ln16] = C[ti][tj][v];
}

// ---------------------------------------------------------------------------
// K3: LayerNorm(p) + output linear (WMMA) + sigmoid gate + residual add.
// ---------------------------------------------------------------------------
__global__ __launch_bounds__(128) void k3_out(
    const float* __restrict__ p_t, const float* __restrict__ gsig,
    const float* __restrict__ pair,
    const float* __restrict__ ng, const float* __restrict__ nb,
    const bf16_t* __restrict__ Wob, const float* __restrict__ b_o,
    float* __restrict__ out, int tiles_per_wave)
{
    const int lane = threadIdx.x & 31;
    const int wave = threadIdx.x >> 5;
    const int h = lane >> 4, ln16 = lane & 15;

    for (int t = 0; t < tiles_per_wave; ++t) {
        const int tile    = (blockIdx.x * 4 + wave) * tiles_per_wave + t;
        const int rowbase = tile * 16;
        const int row     = rowbase + ln16;

        float xv[64];
        float s = 0.0f, ss = 0.0f;
#pragma unroll
        for (int g = 0; g < 8; ++g) {
            const int db = (g >> 1) * 32 + (g & 1) * 16 + h * 8;
#pragma unroll
            for (int e = 0; e < 8; ++e) {
                float x = p_t[(size_t)(db + e) * NROW + row];
                xv[g * 8 + e] = x;
                s += x; ss += x * x;
            }
        }
        s  += __shfl_xor(s, 16);
        ss += __shfl_xor(ss, 16);
        const float mu   = s * (1.0f / 128.0f);
        const float var  = ss * (1.0f / 128.0f) - mu * mu;
        const float rstd = rsqrtf(var + EPSV);

        v16bf A[4];
#pragma unroll
        for (int g = 0; g < 8; ++g) {
            const int db = (g >> 1) * 32 + (g & 1) * 16 + h * 8;
            v4f g0 = *(const v4f*)(ng + db);
            v4f g1 = *(const v4f*)(ng + db + 4);
            v4f c0 = *(const v4f*)(nb + db);
            v4f c1 = *(const v4f*)(nb + db + 4);
#pragma unroll
            for (int e = 0; e < 4; ++e) {
                float n0 = (xv[g * 8 + e]     - mu) * rstd * g0[e] + c0[e];
                float n1 = (xv[g * 8 + 4 + e] - mu) * rstd * g1[e] + c1[e];
                A[g >> 1][(g & 1) * 8 + e]     = (bf16_t)n0;
                A[g >> 1][(g & 1) * 8 + 4 + e] = (bf16_t)n1;
            }
        }

#pragma unroll
        for (int nt = 0; nt < 8; ++nt) {
            v8f c = zero8();
#pragma unroll
            for (int ks = 0; ks < 4; ++ks)
                c = wmma_bf16(A[ks], load_bfrag_w(Wob, nt, ks, lane), c);
            const float bb = b_o[nt * 16 + ln16];
#pragma unroll
            for (int v = 0; v < 8; ++v) {
                const size_t ridx =
                    (size_t)(rowbase + v + 8 * h) * DD + nt * 16 + ln16;
                out[ridx] = pair[ridx] + gsig[ridx] * (c[v] + bb);
            }
        }
    }
}

// ---------------------------------------------------------------------------
// Host launcher
// ---------------------------------------------------------------------------
extern "C" void kernel_launch(void* const* d_in, const int* in_sizes, int n_in,
                              void* d_out, int out_size, void* d_ws, size_t ws_size,
                              hipStream_t stream) {
    const float* pair   = (const float*)d_in[0];
    const float* norm_g = (const float*)d_in[1];
    const float* norm_b = (const float*)d_in[2];
    const float* W_ga = (const float*)d_in[3];  const float* b_ga = (const float*)d_in[4];
    const float* W_la = (const float*)d_in[5];  const float* b_la = (const float*)d_in[6];
    const float* W_gb = (const float*)d_in[7];  const float* b_gb = (const float*)d_in[8];
    const float* W_lb = (const float*)d_in[9];  const float* b_lb = (const float*)d_in[10];
    const float* W_g  = (const float*)d_in[11]; const float* b_g  = (const float*)d_in[12];
    const float* nout_g = (const float*)d_in[13]; const float* nout_b = (const float*)d_in[14];
    const float* W_o  = (const float*)d_in[15]; const float* b_o  = (const float*)d_in[16];

    char* ws = (char*)d_ws;
    size_t off = 0;
    bf16_t* Wb = (bf16_t*)(ws + off);  off += (size_t)6 * DD * DD * sizeof(bf16_t);
    off = (off + 255) & ~(size_t)255;
    bf16_t* la_t = (bf16_t*)(ws + off); off += (size_t)DD * NROW * sizeof(bf16_t);  // 64MB
    bf16_t* lb_t = (bf16_t*)(ws + off); off += (size_t)DD * NROW * sizeof(bf16_t);  // 64MB
    float*  p_t  = (float*)(ws + off);  off += (size_t)DD * NROW * sizeof(float);   // 128MB
    float*  gsig = (float*)(ws + off);  off += (size_t)NROW * DD * sizeof(float);   // 128MB
    if (off > ws_size) return;  // deterministic no-op if scratch insufficient

    k0_cvt_weights<<<(6 * DD * DD + 255) / 256, 256, 0, stream>>>(
        W_ga, W_la, W_gb, W_lb, W_g, W_o, Wb);

    k1_ln_proj<<<1024, 128, 0, stream>>>(
        pair, norm_g, norm_b, Wb, b_ga, b_la, b_gb, b_lb, b_g,
        la_t, lb_t, gsig, 4);

    dim3 g2(4, 4, 128);
    k2_triangle<<<g2, 256, 0, stream>>>(la_t, lb_t, p_t);

    k3_out<<<1024, 128, 0, stream>>>(
        p_t, gsig, pair, nout_g, nout_b, Wb + (size_t)5 * DD * DD, b_o,
        (float*)d_out, 4);
}
